// LlamaAttention_22943715295456
// MI455X (gfx1250) — matrix operand order repaired and checked by reading, not compiled, not run on hardware
//
#include <hip/hip_runtime.h>
#include <hip/hip_bf16.h>

// ---------------------------------------------------------------------------
// Llama attention block for MI455X (gfx1250, wave32).
//   - all GEMMs + flash attention on v_wmma_f32_16x16x32_bf16 (f32 accum)
//   - GEMM tiles staged by the Tensor Data Mover (tensor_load_to_lds) with
//     double-buffered LDS and s_wait_tensorcnt pipelining
// ---------------------------------------------------------------------------

typedef unsigned short u16;
typedef unsigned int   u32;
typedef unsigned long long u64;
typedef __attribute__((ext_vector_type(4)))  u32    v4u;
typedef __attribute__((ext_vector_type(8)))  u32    v8u;
typedef __attribute__((ext_vector_type(8)))  float  v8f;
typedef __attribute__((ext_vector_type(8)))  __bf16 v8bf;
typedef __attribute__((ext_vector_type(16))) __bf16 v16bf;

constexpr int B_   = 2;
constexpr int S_   = 2048;
constexpr int HID_ = 2048;
constexpr int NH_  = 16;
constexpr int NKV_ = 4;
constexpr int HD_  = 128;
constexpr int QW   = NH_ * HD_;    // 2048
constexpr int KVW  = NKV_ * HD_;   // 512
constexpr int MROWS = B_ * S_;     // 4096

__device__ __forceinline__ u16 f2bf(float f) {
  unsigned u = __float_as_uint(f);
  u += 0x7FFFu + ((u >> 16) & 1u);           // round-to-nearest-even
  return (u16)(u >> 16);
}
__device__ __forceinline__ float bf2f(u16 h) {
  return __uint_as_float(((unsigned)h) << 16);
}
__device__ __forceinline__ v16bf cat8(v8bf lo, v8bf hi) {
  return __builtin_shufflevector(lo, hi, 0,1,2,3,4,5,6,7,8,9,10,11,12,13,14,15);
}

__device__ __forceinline__ void wait_tensorcnt0() {
#if __has_builtin(__builtin_amdgcn_s_wait_tensorcnt)
  __builtin_amdgcn_s_wait_tensorcnt(0);
#else
  asm volatile("s_wait_tensorcnt 0x0" ::: "memory");
#endif
}

// ---------------------------------------------------------------------------
// TDM: DMA a 2-D bf16 tile [rows x 32] (row stride = stride_elems) from
// global memory into LDS at lds_off, row-major contiguous.
// D# encoding per CDNA5 ISA 8.3/8.4: group0 = {count/type/lds/global addr},
// group1 = {data_size, tensor dims, tile dims, strides}. Groups 2/3 zero
// (2-D tensor: dim2/dim3 unused).
// ---------------------------------------------------------------------------
__device__ __forceinline__ void tdm_load_tile_2d(const u16* gptr, u32 lds_off,
                                                 u32 rows, u32 tensor_rows,
                                                 u32 stride_elems) {
  const u64 ga = (u64)(uintptr_t)gptr;
  v4u g0;
  g0[0] = 1u;                                         // count=1, user mode
  g0[1] = lds_off;                                    // LDS byte address
  g0[2] = (u32)ga;                                    // global addr [31:0]
  g0[3] = ((u32)(ga >> 32) & 0x01FFFFFFu) | (2u << 30); // addr[56:32], type=2

  const u32 td0 = stride_elems;                       // tensor_dim0 (elems)
  const u32 td1 = tensor_rows;                        // tensor_dim1 (rows)
  v8u g1;
  g1[0] = 1u << 16;                                   // data_size = 2 bytes
  g1[1] = (td0 & 0xFFFFu) << 16;                      // tensor_dim0 lo
  g1[2] = (td0 >> 16) | ((td1 & 0xFFFFu) << 16);      // dim0 hi | dim1 lo
  g1[3] = (td1 >> 16) | (32u << 16);                  // dim1 hi | tile_dim0=32
  g1[4] = rows & 0xFFFFu;                             // tile_dim1, tile_dim2=0
  g1[5] = stride_elems;                               // tensor_dim0_stride lo
  g1[6] = 0u;                                         // stride hi, dim1_stride
  g1[7] = 0u;

  const v4u gz = {0u, 0u, 0u, 0u};
  asm volatile("tensor_load_to_lds %0, %1, %2, %3"
               :: "s"(g0), "s"(g1), "s"(gz), "s"(gz)
               : "memory");
}

// ---------------------------------------------------------------------------
// fp32 -> bf16 conversion
// ---------------------------------------------------------------------------
__global__ __launch_bounds__(256)
void cvt_f32_bf16(const float* __restrict__ in, u16* __restrict__ out, int n) {
  int i = blockIdx.x * 256 + threadIdx.x;
  if (i < n) out[i] = f2bf(in[i]);
}

// ---------------------------------------------------------------------------
// Tiled GEMM:  C[M,N] = A[M,K] (row-major bf16)  x  W[N,K]^T (row-major bf16)
// Block tile 64(M) x 128(N), K-step 32, TDM-staged double-buffered LDS,
// 8 waves x 4 WMMA tiles.
// ---------------------------------------------------------------------------
template <bool OUT_F32>
__global__ __launch_bounds__(256)
void gemm_bf16(const u16* __restrict__ A, const u16* __restrict__ W,
               void* __restrict__ Cout, int M, int N, int K) {
  __shared__ alignas(32) u16 As[2][64 * 32];    // A tile  [m][k]
  __shared__ alignas(32) u16 Bs[2][128 * 32];   // W tile  [n][k]

  const int tid   = threadIdx.x;
  const int lane  = tid & 31;
  const int wave  = tid >> 5;                // 0..7
  const int msub  = wave & 3;                // 4 M-strips of 16
  const int nhalf = wave >> 2;               // 2 N-halves of 64
  const int bm    = blockIdx.y * 64;
  const int bn    = blockIdx.x * 128;

  const int lmod  = lane & 15;
  const int lhalf = lane >> 4;
  const int koff  = lhalf * 8;               // A-fragment K sub-offset
  const int kb16  = lhalf * 16;              // unused; B fragment also uses koff

  const u16* Ablk = A + (size_t)bm * K;
  const u16* Wblk = W + (size_t)bn * K;
  const u32 lAs[2] = {(u32)(uintptr_t)&As[0][0], (u32)(uintptr_t)&As[1][0]};
  const u32 lBs[2] = {(u32)(uintptr_t)&Bs[0][0], (u32)(uintptr_t)&Bs[1][0]};

  v8f acc[4] = {};

  // prologue: DMA first K-slab into buffer 0 (one wave drives the TDM)
  if (wave == 0) {
    tdm_load_tile_2d(Ablk, lAs[0], 64u, (u32)M, (u32)K);
    tdm_load_tile_2d(Wblk, lBs[0], 128u, (u32)N, (u32)K);
  }

  int cur = 0;
  for (int k0 = 0; k0 < K; k0 += 32, cur ^= 1) {
    if (wave == 0) wait_tensorcnt0();        // current tiles landed in LDS
    __syncthreads();
    if (wave == 0 && k0 + 32 < K) {          // overlap: DMA next K-slab
      tdm_load_tile_2d(Ablk + k0 + 32, lAs[cur ^ 1], 64u, (u32)M, (u32)K);
      tdm_load_tile_2d(Wblk + k0 + 32, lBs[cur ^ 1], 128u, (u32)N, (u32)K);
    }

    const int mrow = msub * 16 + lmod;
    v8bf alo = *(const v8bf*)&As[cur][mrow * 32 + koff];
    v8bf ahi = *(const v8bf*)&As[cur][mrow * 32 + koff + 16];
    v16bf afrag = cat8(alo, ahi);

#pragma unroll
    for (int t = 0; t < 4; ++t) {
      const int nrow = nhalf * 64 + t * 16 + lmod;
      v16bf bfrag = cat8(*(const v8bf*)&Bs[cur][nrow * 32 + koff],
                         *(const v8bf*)&Bs[cur][nrow * 32 + koff + 16]);
      acc[t] = __builtin_amdgcn_wmma_f32_16x16x32_bf16(
          false, afrag, false, bfrag, (short)0, acc[t], false, false);
    }
    __syncthreads();                         // buffer free before reuse
  }

  // C layout: VGPR i -> row i + 8*(lane>>4); col = lane&15
#pragma unroll
  for (int t = 0; t < 4; ++t) {
#pragma unroll
    for (int i = 0; i < 8; ++i) {
      const int r = bm + msub * 16 + i + 8 * lhalf;
      const int c = bn + nhalf * 64 + t * 16 + lmod;
      if (OUT_F32)
        ((float*)Cout)[(size_t)r * N + c] = acc[t][i];
      else
        ((u16*)Cout)[(size_t)r * N + c] = f2bf(acc[t][i]);
    }
  }
}

// ---------------------------------------------------------------------------
// RoPE (in-place, bf16): d in [0,64) paired with d+64, theta = 10000.
// ---------------------------------------------------------------------------
__global__ __launch_bounds__(256)
void rope_kernel(u16* __restrict__ X, const int* __restrict__ pos_ids,
                 int nheads) {
  const int idx = blockIdx.x * 256 + threadIdx.x;
  const int d = idx & 63;
  const int h = (idx >> 6) % nheads;
  const int s = (idx / (64 * nheads)) % S_;
  const int b = idx / (64 * nheads * S_);
  if (b >= B_) return;

  const float pos = (float)pos_ids[b * S_ + s];
  const float inv = __expf(-(float)d * (logf(10000.0f) / 64.0f));
  float sn, cs;
  __sincosf(pos * inv, &sn, &cs);

  const size_t off = (size_t)(b * S_ + s) * (nheads * HD_) + h * HD_ + d;
  const float lo = bf2f(X[off]);
  const float hi = bf2f(X[off + 64]);
  X[off]      = f2bf(lo * cs - hi * sn);
  X[off + 64] = f2bf(hi * cs + lo * sn);
}

// ---------------------------------------------------------------------------
// Flash attention (causal, GQA 4:1). Block = 64 q-rows of one (b,h);
// 4 waves, each owning a 16-row Q tile and a 16x128 f32 accumulator.
// Per 32-key chunk: 8 score WMMAs + 8 PV WMMAs per wave.
// ---------------------------------------------------------------------------
__global__ __launch_bounds__(128)
void attn_kernel(const u16* __restrict__ Qb, const u16* __restrict__ Kb,
                 const u16* __restrict__ Vb, u16* __restrict__ Ob) {
  __shared__ alignas(32) u16 Vt[128 * 32];      // V^T chunk: [d][kv]
  __shared__ alignas(32) u16 Pls[4][16 * 32];   // per-wave P staging

  const int tid  = threadIdx.x;
  const int lane = tid & 31;
  const int wave = tid >> 5;
  const int lmod = lane & 15, lhalf = lane >> 4;
  const int koff = lhalf * 8, kb16 = lhalf * 16;

  const int h   = blockIdx.y;
  const int b   = blockIdx.z;
  const int kvh = h >> 2;                        // NH/NKV = 4
  const int q0b = blockIdx.x * 64;
  const int q0  = q0b + wave * 16;
  const int bS  = b * S_;
  const float SSCALE = 0.08838834764831843f;    // 1/sqrt(128)

  // Q fragments (A-matrix role), resident for the whole kernel
  const u16* Qrow = Qb + (size_t)(bS + q0 + lmod) * QW + h * HD_;
  v16bf qfrag[4];
#pragma unroll
  for (int c = 0; c < 4; ++c) {
    v8bf lo = *(const v8bf*)(Qrow + c * 32 + koff);
    v8bf hi = *(const v8bf*)(Qrow + c * 32 + koff + 16);
    qfrag[c] = cat8(lo, hi);
  }

  v8f acc[8] = {};
  float mstat[8], lstat[8];
#pragma unroll
  for (int i = 0; i < 8; ++i) { mstat[i] = -1e30f; lstat[i] = 0.0f; }

  auto score16 = [&](int kvt, float sv[8]) {
    const u16* kptr = Kb + (size_t)(bS + kvt + lmod) * KVW + kvh * HD_;
    v8f s = {};
#pragma unroll
    for (int c = 0; c < 4; ++c) {
      v16bf kf = cat8(*(const v8bf*)(kptr + c * 32 + koff),
                      *(const v8bf*)(kptr + c * 32 + koff + 16));
      s = __builtin_amdgcn_wmma_f32_16x16x32_bf16(
          false, qfrag[c], false, kf, (short)0, s, false, false);
    }
#pragma unroll
    for (int i = 0; i < 8; ++i) {
      const int col = kvt + lmod;
      const int row = q0 + i + 8 * lhalf;
      sv[i] = (col > row) ? -1e30f : s[i] * SSCALE;
    }
  };

  const int nchunk = (q0b + 64) >> 5;
  for (int ch = 0; ch < nchunk; ++ch) {
    const int kv0 = ch << 5;

    // --- cooperative V^T staging: thread = (kv row, 32-wide d segment) ---
    {
      const int kv = tid & 31, dseg = tid >> 5;
      const u16* vrow =
          Vb + (size_t)(bS + kv0 + kv) * KVW + kvh * HD_ + dseg * 32;
      u16 tmp[32];
      *(uint4*)&tmp[0]  = *(const uint4*)&vrow[0];
      *(uint4*)&tmp[8]  = *(const uint4*)&vrow[8];
      *(uint4*)&tmp[16] = *(const uint4*)&vrow[16];
      *(uint4*)&tmp[24] = *(const uint4*)&vrow[24];
#pragma unroll
      for (int j = 0; j < 32; ++j) Vt[(dseg * 32 + j) * 32 + kv] = tmp[j];
    }
    __syncthreads();

    if (kv0 <= q0 + 15) {
      float s0[8], s1[8];
      score16(kv0, s0);
      if (kv0 + 16 <= q0 + 15) {
        score16(kv0 + 16, s1);
      } else {
#pragma unroll
        for (int i = 0; i < 8; ++i) s1[i] = -1e30f;
      }

      // --- online softmax (row reductions across the 16-lane half) ---
#pragma unroll
      for (int i = 0; i < 8; ++i) {
        float t = fmaxf(s0[i], s1[i]);
#pragma unroll
        for (int m = 1; m < 16; m <<= 1) t = fmaxf(t, __shfl_xor(t, m, 32));
        const float nm = fmaxf(mstat[i], t);
        const float sc = __expf(mstat[i] - nm);
        const float p0 = __expf(s0[i] - nm);
        const float p1 = __expf(s1[i] - nm);
        float rs = p0 + p1;
#pragma unroll
        for (int m = 1; m < 16; m <<= 1) rs += __shfl_xor(rs, m, 32);
        lstat[i] = lstat[i] * sc + rs;
        mstat[i] = nm;
        s0[i] = p0;
        s1[i] = p1;
#pragma unroll
        for (int dt = 0; dt < 8; ++dt) acc[dt][i] *= sc;
      }

      // --- P: C-layout -> A-layout via per-wave LDS round-trip ---
      u16* P = &Pls[wave][0];
#pragma unroll
      for (int i = 0; i < 8; ++i) {
        const int r = i + 8 * lhalf;
        P[r * 32 + lmod]      = f2bf(s0[i]);
        P[r * 32 + 16 + lmod] = f2bf(s1[i]);
      }
      v8bf plo = *(const v8bf*)&P[lmod * 32 + koff];
      v8bf phi = *(const v8bf*)&P[lmod * 32 + koff + 16];
      v16bf pfrag = cat8(plo, phi);

      // --- P x V: 8 d-tiles of 16 ---
#pragma unroll
      for (int dt = 0; dt < 8; ++dt) {
        v16bf vf = cat8(*(const v8bf*)&Vt[(dt * 16 + lmod) * 32 + koff],
                        *(const v8bf*)&Vt[(dt * 16 + lmod) * 32 + koff + 16]);
        acc[dt] = __builtin_amdgcn_wmma_f32_16x16x32_bf16(
            false, pfrag, false, vf, (short)0, acc[dt], false, false);
      }
    }
    __syncthreads();
  }

  // epilogue: normalize and store bf16 into (B*S, NH*HD) row-major
#pragma unroll
  for (int dt = 0; dt < 8; ++dt) {
#pragma unroll
    for (int i = 0; i < 8; ++i) {
      const int r = i + 8 * lhalf;
      const int s = q0 + r;
      Ob[(size_t)(bS + s) * QW + h * HD_ + dt * 16 + lmod] =
          f2bf(acc[dt][i] / lstat[i]);
    }
  }
}

// ---------------------------------------------------------------------------
// host launcher
// ---------------------------------------------------------------------------
extern "C" void kernel_launch(void* const* d_in, const int* in_sizes, int n_in,
                              void* d_out, int out_size, void* d_ws,
                              size_t ws_size, hipStream_t stream) {
  (void)in_sizes; (void)n_in; (void)out_size; (void)ws_size;
  const float* hs = (const float*)d_in[0];
  const float* Wq = (const float*)d_in[1];
  const float* Wk = (const float*)d_in[2];
  const float* Wv = (const float*)d_in[3];
  const float* Wo = (const float*)d_in[4];
  const int*  pos = (const int*)d_in[5];
  float* out = (float*)d_out;

  // workspace layout (bf16 elements)
  u16* ws  = (u16*)d_ws;
  u16* Xb  = ws;                           // 4096x2048
  u16* Wqb = Xb  + (size_t)MROWS * HID_;   // 2048x2048
  u16* Wkb = Wqb + (size_t)QW * HID_;      //  512x2048
  u16* Wvb = Wkb + (size_t)KVW * HID_;     //  512x2048
  u16* Wob = Wvb + (size_t)KVW * HID_;     // 2048x2048
  u16* Qb  = Wob + (size_t)HID_ * QW;      // 4096x2048
  u16* Kb  = Qb  + (size_t)MROWS * QW;     // 4096x512
  u16* Vb  = Kb  + (size_t)MROWS * KVW;    // 4096x512
  u16* Ab  = Vb  + (size_t)MROWS * KVW;    // 4096x2048

  auto cvt = [&](const float* src, u16* dst, int n) {
    cvt_f32_bf16<<<(n + 255) / 256, 256, 0, stream>>>(src, dst, n);
  };
  cvt(hs, Xb, MROWS * HID_);
  cvt(Wq, Wqb, QW * HID_);
  cvt(Wk, Wkb, KVW * HID_);
  cvt(Wv, Wvb, KVW * HID_);
  cvt(Wo, Wob, HID_ * QW);

  // QKV projections
  gemm_bf16<false><<<dim3(QW / 128, MROWS / 64), 256, 0, stream>>>(
      Xb, Wqb, Qb, MROWS, QW, HID_);
  gemm_bf16<false><<<dim3(KVW / 128, MROWS / 64), 256, 0, stream>>>(
      Xb, Wkb, Kb, MROWS, KVW, HID_);
  gemm_bf16<false><<<dim3(KVW / 128, MROWS / 64), 256, 0, stream>>>(
      Xb, Wvb, Vb, MROWS, KVW, HID_);

  // RoPE
  rope_kernel<<<(B_ * S_ * NH_ * 64) / 256, 256, 0, stream>>>(Qb, pos, NH_);
  rope_kernel<<<(B_ * S_ * NKV_ * 64) / 256, 256, 0, stream>>>(Kb, pos, NKV_);

  // flash attention
  attn_kernel<<<dim3(S_ / 64, NH_, B_), 128, 0, stream>>>(Qb, Kb, Vb, Ab);

  // output projection -> f32 d_out
  gemm_bf16<true><<<dim3(HID_ / 128, MROWS / 64), 256, 0, stream>>>(
      Ab, Wob, out, MROWS, HID_, QW);
}
